// GraphResNet_75771813036513
// MI455X (gfx1250) — compile-verified, matrix operand order checked by reading, not run on hardware
//
#include <hip/hip_runtime.h>

// ---------------------------------------------------------------------------
// Types for CDNA5 WMMA (gfx1250, wave32)
// ---------------------------------------------------------------------------
typedef __attribute__((ext_vector_type(8)))  float  v8f;
typedef __attribute__((ext_vector_type(16))) __bf16 v16bf;
typedef __attribute__((ext_vector_type(8)))  __bf16 v8bf;

__device__ __forceinline__ unsigned short f2bf_bits(float f) {
    unsigned u = __builtin_bit_cast(unsigned, f);
    u += 0x7fffu + ((u >> 16) & 1u);          // round-to-nearest-even
    return (unsigned short)(u >> 16);
}

// Load one 16x32 bf16 WMMA fragment (ISA 16-bit A/B layout): the lane's 16
// elements are K = koff..koff+7 and K = 16+koff..16+koff+7, already folded
// into the pointer by the caller; two 16-byte loads.
__device__ __forceinline__ v16bf ldfrag(const unsigned short* p) {
    v8bf x0 = *(const v8bf*)p;
    v8bf x1 = *(const v8bf*)(p + 16);
    return __builtin_shufflevector(x0, x1, 0,1,2,3,4,5,6,7,8,9,10,11,12,13,14,15);
}

// ---------------------------------------------------------------------------
// Elementwise / utility kernels
// ---------------------------------------------------------------------------
__global__ void fill_f32(float* p, float v, int n) {
    int i = blockIdx.x * blockDim.x + threadIdx.x;
    if (i < n) p[i] = v;
}

__global__ void deg_count(const int* __restrict__ dst, float* __restrict__ deg, int E) {
    int e = blockIdx.x * blockDim.x + threadIdx.x;
    if (e < E) unsafeAtomicAdd(&deg[dst[e]], 1.0f);
}

__global__ void rsqrt_inplace(float* p, int n) {
    int i = blockIdx.x * blockDim.x + threadIdx.x;
    if (i < n) p[i] = rsqrtf(p[i]);
}

__global__ void f32_to_bf16(const float* __restrict__ in, unsigned short* __restrict__ out, int n) {
    int i = blockIdx.x * blockDim.x + threadIdx.x;
    if (i < n) out[i] = f2bf_bits(in[i]);
}

// wt[m*K + k] = bf16(w[k*M + m])   (w is [K x M] row-major)
__global__ void transpose_to_bf16(const float* __restrict__ w, unsigned short* __restrict__ wt,
                                  int K, int M) {
    int i = blockIdx.x * blockDim.x + threadIdx.x;
    if (i >= K * M) return;
    int k = i / M, m = i % M;
    wt[(size_t)m * K + k] = f2bf_bits(w[(size_t)k * M + m]);
}

__global__ void relu_inplace(float* p, int n) {
    int i = blockIdx.x * blockDim.x + threadIdx.x;
    if (i < n) p[i] = fmaxf(p[i], 0.0f);
}

// h = relu(u) + r
__global__ void add_relu(float* __restrict__ h, const float* __restrict__ u,
                         const float* __restrict__ r, int n) {
    int i = blockIdx.x * blockDim.x + threadIdx.x;
    if (i < n) h[i] = fmaxf(u[i], 0.0f) + r[i];
}

// ---------------------------------------------------------------------------
// WMMA bf16 GEMM:  out[rows x M] = A[rows x K] * B[K x M] (+bias)(+relu)
// A bf16 row-major; B transposed as BT[M x K] bf16 row-major.
// Register-blocked: each wave owns a 32x64 macro-tile (2 M-tiles x 4 N-tiles,
// 8 f32 accumulators). Per k-step: 4 A-loads + 8 B-loads feed 8 WMMAs
// (1.5 b128 loads per v_wmma vs 4 for a 16x16-per-wave tiling).
// Block = 64 threads (2 waves, adjacent 64-col groups).
// grid = (M/128, rows/32). Requires rows%32==0, M%128==0, K%32==0.
// ---------------------------------------------------------------------------
__global__ void wmma_gemm_bf16(const unsigned short* __restrict__ A,
                               const unsigned short* __restrict__ BT,
                               const float* __restrict__ bias,   // may be null
                               float* __restrict__ out,
                               int rows, int K, int M, int relu) {
    const int lane    = threadIdx.x & 31;
    const int wave    = threadIdx.x >> 5;
    const int colBase = (blockIdx.x * 2 + wave) * 64;
    const int rowBase = blockIdx.y * 32;
    const int l15     = lane & 15;
    const int koff    = (lane >> 4) * 8;

    const unsigned short* ap[2];
    ap[0] = A + (size_t)(rowBase + l15) * K + koff;
    ap[1] = ap[0] + (size_t)16 * K;
    const unsigned short* bp[4];
    #pragma unroll
    for (int j = 0; j < 4; ++j)
        bp[j] = BT + (size_t)(colBase + j * 16 + l15) * K + koff;

    v8f acc[2][4] = {};
    for (int k = 0; k < K; k += 32) {
        v16bf a[2], b[4];
        #pragma unroll
        for (int i = 0; i < 2; ++i) a[i] = ldfrag(ap[i] + k);
        #pragma unroll
        for (int j = 0; j < 4; ++j) b[j] = ldfrag(bp[j] + k);
        #pragma unroll
        for (int i = 0; i < 2; ++i)
            #pragma unroll
            for (int j = 0; j < 4; ++j)
                acc[i][j] = __builtin_amdgcn_wmma_f32_16x16x32_bf16(
                                false, a[i], false, b[j], (short)0, acc[i][j],
                                false, false);
    }

    // Epilogue: D layout — lane holds column (lane&15) of the tile for rows
    // M = 8*(lane>>4) + r, r = 0..7.
    #pragma unroll
    for (int j = 0; j < 4; ++j) {
        const int col = colBase + j * 16 + l15;
        const float bb = bias ? bias[col] : 0.0f;
        #pragma unroll
        for (int i = 0; i < 2; ++i) {
            const int mbase = rowBase + i * 16 + ((lane >> 4) << 3);
            float* op = out + (size_t)mbase * M + col;
            #pragma unroll
            for (int r = 0; r < 8; ++r) {
                float v = acc[i][j][r] + bb;
                if (relu) v = fmaxf(v, 0.0f);
                op[(size_t)r * M] = v;
            }
        }
    }
}

// ---------------------------------------------------------------------------
// GCN sparse part:  out = dinv*Agg(dinv[s]*xw[s]) + dinv^2*xw + b
// gcn_init seeds out with the self-loop + bias term; gcn_scatter atomically
// accumulates edge messages (both dinv factors folded in per edge).
// ---------------------------------------------------------------------------
__global__ void gcn_init(float* __restrict__ out, const float* __restrict__ xw,
                         const float* __restrict__ dinv, const float* __restrict__ bias,
                         int N, int C) {
    int i = blockIdx.x * blockDim.x + threadIdx.x;
    if (i >= N * C) return;
    int row = i / C, c = i % C;
    float di = dinv[row];
    out[i] = xw[i] * di * di + bias[c];
}

__global__ void gcn_scatter(float* __restrict__ out, const float* __restrict__ xw,
                            const int* __restrict__ src, const int* __restrict__ dst,
                            const float* __restrict__ dinv, int E, int C) {
    int cq = C >> 2;  // float4 chunks per row
    int t = blockIdx.x * blockDim.x + threadIdx.x;
    if (t >= E * cq) return;
    int e  = t / cq;
    int c4 = (t % cq) << 2;
    int s = src[e], d = dst[e];
    float w = dinv[s] * dinv[d];
    const float4 v = *(const float4*)(xw + (size_t)s * C + c4);
    float* o = out + (size_t)d * C + c4;
    unsafeAtomicAdd(o + 0, v.x * w);
    unsafeAtomicAdd(o + 1, v.y * w);
    unsafeAtomicAdd(o + 2, v.z * w);
    unsafeAtomicAdd(o + 3, v.w * w);
}

// ---------------------------------------------------------------------------
// BatchNorm (training-mode, biased stats over N nodes) + ReLU
// ---------------------------------------------------------------------------
__global__ void bn_stats(const float* __restrict__ x, float* __restrict__ sum,
                         float* __restrict__ sq, int N, int C, int strip) {
    int r0 = blockIdx.x * strip;
    int r1 = min(r0 + strip, N);
    for (int c = threadIdx.x; c < C; c += blockDim.x) {
        float s = 0.f, q = 0.f;
        for (int i = r0; i < r1; ++i) {
            float v = x[(size_t)i * C + c];
            s += v; q += v * v;
        }
        unsafeAtomicAdd(&sum[c], s);
        unsafeAtomicAdd(&sq[c], q);
    }
}

__global__ void bn_apply_relu(float* __restrict__ x, const float* __restrict__ sum,
                              const float* __restrict__ sq, const float* __restrict__ g,
                              const float* __restrict__ b, int N, int C) {
    int i = blockIdx.x * blockDim.x + threadIdx.x;
    if (i >= N * C) return;
    int c = i % C;
    float inv_n = 1.0f / (float)N;
    float m  = sum[c] * inv_n;
    float vv = sq[c] * inv_n - m * m;
    float y  = g[c] * (x[i] - m) * rsqrtf(vv + 1e-5f) + b[c];
    x[i] = fmaxf(y, 0.0f);
}

// ---------------------------------------------------------------------------
// Global mean pool over graphs
// ---------------------------------------------------------------------------
__global__ void pool_cnt(const int* __restrict__ batch, float* __restrict__ cnt, int N) {
    int i = blockIdx.x * blockDim.x + threadIdx.x;
    if (i < N) unsafeAtomicAdd(&cnt[batch[i]], 1.0f);
}

__global__ void pool_sum(const float* __restrict__ h, const int* __restrict__ batch,
                         float* __restrict__ pooled, int N, int C) {
    int i = blockIdx.x * blockDim.x + threadIdx.x;
    if (i >= N * C) return;
    int row = i / C, c = i % C;
    unsafeAtomicAdd(&pooled[(size_t)batch[row] * C + c], h[i]);
}

__global__ void pool_div(float* __restrict__ pooled, const float* __restrict__ cnt,
                         int G, int C) {
    int i = blockIdx.x * blockDim.x + threadIdx.x;
    if (i >= G * C) return;
    pooled[i] /= fmaxf(cnt[i / C], 1.0f);
}

// ---------------------------------------------------------------------------
// Host side
// ---------------------------------------------------------------------------
static inline int cdiv(long long a, long long b) { return (int)((a + b - 1) / b); }

extern "C" void kernel_launch(void* const* d_in, const int* in_sizes, int n_in,
                              void* d_out, int out_size, void* d_ws, size_t ws_size,
                              hipStream_t stream) {
    const float* x_in  = (const float*)d_in[0];
    const int*   src   = (const int*)d_in[1];
    const int*   batch = (const int*)d_in[2];

    const int E = in_sizes[1] / 2;
    const int N = in_sizes[2];
    const int F = in_sizes[0] / N;
    const int* dst = src + E;

    const int C0    = in_sizes[4];
    const int MID0  = in_sizes[6],  COUT0 = in_sizes[10];
    const int MID1  = in_sizes[16], COUT1 = in_sizes[20];
    const int NHID  = in_sizes[26], NOUT  = in_sizes[28];
    const int G     = out_size / NOUT;
    const int CM    = COUT1;   // max width (1024)

    // ---- workspace layout ----
    char* ws = (char*)d_ws;
    size_t off = 0;
    auto alloc = [&](size_t bytes) { char* p = ws + off; off = (off + bytes + 255) & ~(size_t)255; return p; };
    float*          dinv   = (float*)alloc((size_t)N * 4);
    float*          stats  = (float*)alloc((size_t)2 * CM * 4);     // sum | sumsq
    float*          cnt    = (float*)alloc((size_t)G * 4);
    float*          pooled = (float*)alloc((size_t)G * CM * 4);
    unsigned short* wbf    = (unsigned short*)alloc((size_t)CM * CM * 2);
    unsigned short* hbf    = (unsigned short*)alloc((size_t)N * CM * 2);
    float*          bh     = (float*)alloc((size_t)N * CM * 4);     // h
    float*          bt     = (float*)alloc((size_t)N * CM * 4);     // t
    float*          bu     = (float*)alloc((size_t)N * CM * 4);     // u
    float*          xw     = (float*)alloc((size_t)N * CM * 4);
    (void)ws_size; (void)n_in;

    const int TPB = 256;
    auto g1 = [&](long long n) { return dim3(cdiv(n, TPB)); };

    // ---- degrees: deg = 1 + indeg; dinv = rsqrt(deg) ----
    fill_f32<<<g1(N), TPB, 0, stream>>>(dinv, 1.0f, N);
    deg_count<<<g1(E), TPB, 0, stream>>>(dst, dinv, E);
    rsqrt_inplace<<<g1(N), TPB, 0, stream>>>(dinv, N);

    // GEMM launcher: 32x64 macro-tile per wave, 2 waves per block.
    auto gemm = [&](const unsigned short* A, const unsigned short* BT, const float* bias,
                    float* outp, int rows, int K, int M, int relu) {
        wmma_gemm_bf16<<<dim3(M / 128, rows / 32), 64, 0, stream>>>(A, BT, bias, outp,
                                                                    rows, K, M, relu);
    };

    // ---- one GCN layer: out = dinv*Agg(dinv*in@W) + dinv^2*(in@W) + b ----
    auto gcn = [&](const float* in, int K, int M, const float* w, const float* b, float* out) {
        f32_to_bf16<<<g1((long long)N * K), TPB, 0, stream>>>(in, hbf, N * K);
        transpose_to_bf16<<<g1((long long)K * M), TPB, 0, stream>>>(w, wbf, K, M);
        gemm(hbf, wbf, nullptr, xw, N, K, M, 0);
        gcn_init<<<g1((long long)N * M), TPB, 0, stream>>>(out, xw, dinv, b, N, M);
        gcn_scatter<<<g1((long long)E * (M / 4)), TPB, 0, stream>>>(out, xw, src, dst, dinv, E, M);
    };

    auto bn = [&](float* h, const float* g, const float* be, int C) {
        fill_f32<<<g1(2 * C), TPB, 0, stream>>>(stats, 0.0f, 2 * C);
        const int strips = 160;
        bn_stats<<<strips, TPB, 0, stream>>>(h, stats, stats + C, N, C, cdiv(N, strips));
        bn_apply_relu<<<g1((long long)N * C), TPB, 0, stream>>>(h, stats, stats + C, g, be, N, C);
    };

    // ---- stem: h = relu(gcn(x, w0, b0)) ----
    gcn(x_in, F, C0, (const float*)d_in[3], (const float*)d_in[4], bh);
    relu_inplace<<<g1((long long)N * C0), TPB, 0, stream>>>(bh, N * C0);

    // ---- residual blocks ----
    int cins[2]  = { C0,   COUT0 };
    int mids[2]  = { MID0, MID1  };
    int couts[2] = { COUT0, COUT1 };
    for (int i = 0; i < 2; ++i) {
        const int base = 5 + 10 * i;
        const float* w1  = (const float*)d_in[base + 0];
        const float* b1  = (const float*)d_in[base + 1];
        const float* gg1 = (const float*)d_in[base + 2];
        const float* be1 = (const float*)d_in[base + 3];
        const float* w2  = (const float*)d_in[base + 4];
        const float* b2  = (const float*)d_in[base + 5];
        const float* gg2 = (const float*)d_in[base + 6];
        const float* be2 = (const float*)d_in[base + 7];
        const float* w3  = (const float*)d_in[base + 8];
        const float* b3  = (const float*)d_in[base + 9];
        const int cin = cins[i], mid = mids[i], cout = couts[i];

        gcn(bh, cin, mid, w1, b1, bt);      // t = gcn1(h)
        bn(bt, gg1, be1, mid);              // t = relu(bn(t))
        gcn(bt, mid, cout, w2, b2, bu);     // u = gcn2(t)
        bn(bu, gg2, be2, cout);             // u = relu(bn(u))
        gcn(bh, cin, cout, w3, b3, bt);     // t = gcn3(h)   (t is free again)
        add_relu<<<g1((long long)N * cout), TPB, 0, stream>>>(bh, bu, bt, N * cout);
    }

    // ---- global mean pool ----
    fill_f32<<<g1((long long)G * CM), TPB, 0, stream>>>(pooled, 0.0f, G * CM);
    fill_f32<<<g1(G), TPB, 0, stream>>>(cnt, 0.0f, G);
    pool_cnt<<<g1(N), TPB, 0, stream>>>(batch, cnt, N);
    pool_sum<<<g1((long long)N * CM), TPB, 0, stream>>>(bh, batch, pooled, N, CM);
    pool_div<<<g1((long long)G * CM), TPB, 0, stream>>>(pooled, cnt, G, CM);

    // ---- MLP head: out = relu(pooled@mw1 + mb1) @ mw2 + mb2 ----
    const float* mw1 = (const float*)d_in[25];
    const float* mb1 = (const float*)d_in[26];
    const float* mw2 = (const float*)d_in[27];
    const float* mb2 = (const float*)d_in[28];

    f32_to_bf16<<<g1((long long)G * CM), TPB, 0, stream>>>(pooled, hbf, G * CM);
    transpose_to_bf16<<<g1((long long)CM * NHID), TPB, 0, stream>>>(mw1, wbf, CM, NHID);
    gemm(hbf, wbf, mb1, bt, G, CM, NHID, 1);

    f32_to_bf16<<<g1((long long)G * NHID), TPB, 0, stream>>>(bt, hbf, G * NHID);
    transpose_to_bf16<<<g1((long long)NHID * NOUT), TPB, 0, stream>>>(mw2, wbf, NHID, NOUT);
    gemm(hbf, wbf, mb2, (float*)d_out, G, NHID, NOUT, 0);
}
